// ParagraphGAT_59485297050283
// MI455X (gfx1250) — matrix operand order repaired and checked by reading, not compiled, not run on hardware
//
#include <hip/hip_runtime.h>
#include <hip/hip_bf16.h>

// ---------------- problem constants (from reference) ----------------
#define N_NODES 20000
#define N_EDGES 160000
#define EP      (N_EDGES + N_NODES)   // edges + self loops
#define DIM     128
#define NEG_SLOPE 0.2f
#define LN_EPS    1e-5f

static_assert(N_NODES % 16 == 0, "row tiles must cover N exactly");

typedef __attribute__((ext_vector_type(16))) _Float16 v16h;
typedef __attribute__((ext_vector_type(8)))  _Float16 v8h;
typedef __attribute__((ext_vector_type(8)))  float    v8f;

// ---------------- fp32 -> fp16 conversion ----------------
__global__ void pgat_cvt_f16(const float* __restrict__ src,
                             _Float16* __restrict__ dst, int n) {
  int i = blockIdx.x * 256 + threadIdx.x;
  if (i < n) dst[i] = (_Float16)src[i];
}

// ---------------- WMMA GEMM:  Y[n,m] = sum_k A[n,k]*W[m,k] + bias[m] ----------------
// A: [N_NODES, DIM] f16 row-major (activations), W: [M, DIM] f16 row-major
// grid.x = N/16 row tiles, grid.y = M/128; 256 thr = 8 waves, 1 col-tile/wave.
// M is a template constant so the 8 D-matrix row stores become immediate offsets.
template <int M>
__global__ __launch_bounds__(256)
void pgat_gemm_wmma(const _Float16* __restrict__ A,
                    const _Float16* __restrict__ W,
                    const float* __restrict__ bias,
                    float* __restrict__ Y) {
  __shared__ _Float16 As[16 * 136];          // stride 136 halfs: bank-conflict-free b128
  const int t = threadIdx.x;
  const int rowbase = blockIdx.x * 16;

  for (int i = t; i < 16 * DIM; i += 256) {  // stage 16x128 A tile as f16 (no guards: N%16==0)
    int r = i >> 7, k = i & 127;
    As[r * 136 + k] = A[(size_t)(rowbase + r) * DIM + k];
  }
  __syncthreads();

  const int wave = t >> 5, lane = t & 31;
  const int col   = (blockIdx.y * 8 + wave) * 16 + (lane & 15);
  const int rowA  = lane & 15;
  const int kbase = (lane >> 4) * 8;         // A-fragment K base (16-bit A layout)
  const int kbB   = (lane >> 4) * 8;         // B-fragment K base

  v16h afrag[4];                             // preload 4 K-chunks of A from LDS
#pragma unroll
  for (int kc = 0; kc < 4; ++kc) {
    v8h lo = *(const v8h*)&As[rowA * 136 + kc * 32 + kbase];        // K = kc*32+kb..+7
    v8h hi = *(const v8h*)&As[rowA * 136 + kc * 32 + 16 + kbase];   // K = kc*32+16+kb..
#pragma unroll
    for (int j = 0; j < 8; ++j) { afrag[kc][j] = lo[j]; afrag[kc][j + 8] = hi[j]; }
  }

  v8f acc = {};
  const _Float16* wp = W + (size_t)col * DIM + kbB;
#pragma unroll
  for (int kc = 0; kc < 4; ++kc) {
    v8h b0 = *(const v8h*)(wp + kc * 32);
    v8h b1 = *(const v8h*)(wp + kc * 32 + 16);
    v16h bfrag;
#pragma unroll
    for (int j = 0; j < 8; ++j) { bfrag[j] = b0[j]; bfrag[j + 8] = b1[j]; }
    acc = __builtin_amdgcn_wmma_f32_16x16x32_f16(false, afrag[kc], false, bfrag,
                                                 (short)0, acc, false, false);
  }

  // C/D layout: VGPR r -> M = r + 8*(lane>>4), N = lane&15. One base address,
  // 8 stores with compile-time immediate offsets r*M*4.
  const float bv = bias[col];
  float* yp = Y + (size_t)(rowbase + 8 * (lane >> 4)) * M + col;
#pragma unroll
  for (int r = 0; r < 8; ++r) yp[r * M] = acc[r] + bv;
}

// ---------------- per-layer init: segment max/-inf, den/agg zero ----------------
__global__ void pgat_init(float* __restrict__ mmax, float* __restrict__ den,
                          float* __restrict__ agg, int nh) {
  int i = blockIdx.x * 256 + threadIdx.x;
  if (i < nh) { mmax[i] = -__builtin_inff(); den[i] = 0.f; }
  if (i < N_NODES * DIM) agg[i] = 0.f;
}

// ---------------- edge logits + segment max (1 wave / edge) ----------------
template <int H>
__global__ __launch_bounds__(256)
void pgat_edge_logits(const int* __restrict__ ei, const float* __restrict__ xl,
                      const float* __restrict__ xr, const float* __restrict__ att,
                      float* __restrict__ logit, float* __restrict__ mmax) {
  int e = blockIdx.x * 8 + (threadIdx.x >> 5);
  if (e >= EP) return;
  int lane = threadIdx.x & 31;
  int s, d;
  if (e < N_EDGES) { s = ei[e]; d = ei[N_EDGES + e]; } else { s = d = e - N_EDGES; }
  const float* pl = xl + (size_t)s * H * DIM;
  const float* pr = xr + (size_t)d * H * DIM;
#pragma unroll
  for (int h = 0; h < H; ++h) {
    float acc = 0.f;
#pragma unroll
    for (int j = 0; j < 4; ++j) {
      int k = lane + j * 32;
      float v = pl[h * DIM + k] + pr[h * DIM + k];
      v = (v > 0.f) ? v : v * NEG_SLOPE;                 // leaky_relu
      acc += att[h * DIM + k] * v;
    }
    for (int off = 16; off > 0; off >>= 1) acc += __shfl_down(acc, off, 32);
    if (lane == 0) {
      logit[(size_t)e * H + h] = acc;
      __hip_atomic_fetch_max(&mmax[d * H + h], acc,      // global_atomic_max_num_f32
                             __ATOMIC_RELAXED, __HIP_MEMORY_SCOPE_AGENT);
    }
  }
}

// ---------------- exp(logit - max) + segment sum ----------------
template <int H>
__global__ void pgat_edge_exp(const int* __restrict__ ei, float* __restrict__ logit,
                              const float* __restrict__ mmax, float* __restrict__ den) {
  int gid = blockIdx.x * 256 + threadIdx.x;
  if (gid >= EP * H) return;
  int e = gid / H, h = gid - e * H;                      // H is 2^k -> shifts
  int d = (e < N_EDGES) ? ei[N_EDGES + e] : e - N_EDGES;
  float v = __expf(logit[gid] - mmax[d * H + h]);
  logit[gid] = v;                                        // reuse buffer as ex
  atomicAdd(&den[d * H + h], v);                         // global_atomic_add_f32
}

// ---------------- weighted aggregation, mean over heads (1 wave / edge) ----------------
template <int H>
__global__ __launch_bounds__(256)
void pgat_edge_agg(const int* __restrict__ ei, const float* __restrict__ ex,
                   const float* __restrict__ den, const float* __restrict__ xl,
                   float* __restrict__ agg) {
  int e = blockIdx.x * 8 + (threadIdx.x >> 5);
  if (e >= EP) return;
  int lane = threadIdx.x & 31;
  int s, d;
  if (e < N_EDGES) { s = ei[e]; d = ei[N_EDGES + e]; } else { s = d = e - N_EDGES; }
  float a[H];
#pragma unroll
  for (int h = 0; h < H; ++h)
    a[h] = ex[(size_t)e * H + h] / den[d * H + h] * (1.f / H);
  const float* pl = xl + (size_t)s * H * DIM;
#pragma unroll
  for (int j = 0; j < 4; ++j) {
    int k = lane + j * 32;
    float acc = 0.f;
#pragma unroll
    for (int h = 0; h < H; ++h) acc += a[h] * pl[h * DIM + k];
    atomicAdd(&agg[(size_t)d * DIM + k], acc);
  }
}

// ---------------- bias + LayerNorm + residual + ReLU (1 wave / node) ----------------
__global__ __launch_bounds__(256)
void pgat_node_finalize(const float* __restrict__ agg, const float* __restrict__ bias,
                        const float* __restrict__ g, const float* __restrict__ be,
                        const float* __restrict__ res, float* __restrict__ out) {
  int n = blockIdx.x * 8 + (threadIdx.x >> 5);
  if (n >= N_NODES) return;
  int lane = threadIdx.x & 31;
  float v[4], s = 0.f;
#pragma unroll
  for (int j = 0; j < 4; ++j) {
    int k = lane + j * 32;
    v[j] = agg[(size_t)n * DIM + k] + bias[k];
    s += v[j];
  }
  for (int off = 16; off > 0; off >>= 1) s += __shfl_xor(s, off, 32);
  float mu = s * (1.f / DIM), var = 0.f;
#pragma unroll
  for (int j = 0; j < 4; ++j) { float t = v[j] - mu; var += t * t; }
  for (int off = 16; off > 0; off >>= 1) var += __shfl_xor(var, off, 32);
  float rs = rsqrtf(var * (1.f / DIM) + LN_EPS);
#pragma unroll
  for (int j = 0; j < 4; ++j) {
    int k = lane + j * 32;
    float h = (v[j] - mu) * rs * g[k] + be[k] + res[(size_t)n * DIM + k];
    out[(size_t)n * DIM + k] = fmaxf(h, 0.f);
  }
}

// ---------------- host orchestration ----------------
extern "C" void kernel_launch(void* const* d_in, const int* in_sizes, int n_in,
                              void* d_out, int out_size, void* d_ws, size_t ws_size,
                              hipStream_t stream) {
  const float* x  = (const float*)d_in[0];
  const int*   ei = (const int*)d_in[1];

  char* ws = (char*)d_ws;
  size_t off = 0;
  auto carve = [&](size_t bytes) -> char* {
    char* p = ws + off; off += (bytes + 255) & ~(size_t)255; return p;
  };
  _Float16* xh   = (_Float16*)carve((size_t)N_NODES * DIM * 2);
  _Float16* wh   = (_Float16*)carve((size_t)1024 * DIM * 2);
  float* xl      = (float*)carve((size_t)N_NODES * 8 * DIM * 4);
  float* xr      = (float*)carve((size_t)N_NODES * 8 * DIM * 4);
  float* logit   = (float*)carve((size_t)EP * 8 * 4);
  float* mmax    = (float*)carve((size_t)N_NODES * 8 * 4);
  float* den     = (float*)carve((size_t)N_NODES * 8 * 4);
  float* agg     = (float*)carve((size_t)N_NODES * DIM * 4);
  float* hA      = (float*)carve((size_t)N_NODES * DIM * 4);
  float* hB      = (float*)carve((size_t)N_NODES * DIM * 4);

  const int Hs[3] = {8, 8, 4};
  float* houts[3] = {hA, hB, (float*)d_out};
  const float* hin = x;

  const int nTiles  = N_NODES / 16;                  // 1250 row tiles (exact)
  const int eBlocks = (EP + 7) / 8;                  // 1 wave per edge

  for (int l = 0; l < 3; ++l) {
    const float* Wl   = (const float*)d_in[2 + l * 8 + 0];
    const float* bl   = (const float*)d_in[2 + l * 8 + 1];
    const float* Wr   = (const float*)d_in[2 + l * 8 + 2];
    const float* br   = (const float*)d_in[2 + l * 8 + 3];
    const float* att  = (const float*)d_in[2 + l * 8 + 4];
    const float* bias = (const float*)d_in[2 + l * 8 + 5];
    const float* g    = (const float*)d_in[2 + l * 8 + 6];
    const float* be   = (const float*)d_in[2 + l * 8 + 7];
    const int H = Hs[l], M = H * DIM;

    pgat_cvt_f16<<<(N_NODES * DIM + 255) / 256, 256, 0, stream>>>(hin, xh, N_NODES * DIM);

    pgat_cvt_f16<<<(M * DIM + 255) / 256, 256, 0, stream>>>(Wl, wh, M * DIM);
    if (M == 1024) pgat_gemm_wmma<1024><<<dim3(nTiles, 8), 256, 0, stream>>>(xh, wh, bl, xl);
    else           pgat_gemm_wmma< 512><<<dim3(nTiles, 4), 256, 0, stream>>>(xh, wh, bl, xl);
    pgat_cvt_f16<<<(M * DIM + 255) / 256, 256, 0, stream>>>(Wr, wh, M * DIM);
    if (M == 1024) pgat_gemm_wmma<1024><<<dim3(nTiles, 8), 256, 0, stream>>>(xh, wh, br, xr);
    else           pgat_gemm_wmma< 512><<<dim3(nTiles, 4), 256, 0, stream>>>(xh, wh, br, xr);

    pgat_init<<<(N_NODES * DIM + 255) / 256, 256, 0, stream>>>(mmax, den, agg, N_NODES * H);

    if (H == 8) {
      pgat_edge_logits<8><<<eBlocks, 256, 0, stream>>>(ei, xl, xr, att, logit, mmax);
      pgat_edge_exp<8><<<(EP * 8 + 255) / 256, 256, 0, stream>>>(ei, logit, mmax, den);
      pgat_edge_agg<8><<<eBlocks, 256, 0, stream>>>(ei, logit, den, xl, agg);
    } else {
      pgat_edge_logits<4><<<eBlocks, 256, 0, stream>>>(ei, xl, xr, att, logit, mmax);
      pgat_edge_exp<4><<<(EP * 4 + 255) / 256, 256, 0, stream>>>(ei, logit, mmax, den);
      pgat_edge_agg<4><<<eBlocks, 256, 0, stream>>>(ei, logit, den, xl, agg);
    }

    pgat_node_finalize<<<(N_NODES + 7) / 8, 256, 0, stream>>>(agg, bias, g, be, hin, houts[l]);
    hin = houts[l];
  }
}
